// MultiheadAttention_26079041421841
// MI455X (gfx1250) — compile-verified, matrix operand order checked by reading, not run on hardware
//
#include <hip/hip_runtime.h>
#include <math.h>

// MultiheadAttention forward for MI455X (gfx1250).
// bf16 WMMA GEMMs, double-buffered async global->LDS tile staging.
// B=2, S=2048, D_MODEL=1024, H=16, DEPTH=64.
// d_out = [output (B*S*D) f32 | attn (B*H*S*S) f32]
// d_ws  = bf16 copies of activations/weights + bf16 q,k,v,concat (64 MB).

typedef __attribute__((ext_vector_type(16))) __bf16        v16bf;
typedef __attribute__((ext_vector_type(8)))  unsigned int  v8u;
typedef __attribute__((ext_vector_type(8)))  float         v8f;

#define BATCH 2
#define SEQ   2048
#define DM    1024
#define NH    16
#define DH    64
// LDS row strides (bf16 elems). Rows are multiples of 16B (async B128 dest
// alignment); r*20 mod 64 and r*36 mod 64 are distinct for r=0..15 -> the
// 16-row fragment reads are bank-conflict-free.
#define LDSTR  40   // 80 B rows (32-wide K tiles)
#define LDSTR2 72   // 144 B rows (64-wide K tiles)

#define LDS_SPACE __attribute__((address_space(3)))

__device__ __forceinline__ unsigned lds_off(const void* p) {
  return (unsigned)(unsigned long long)(LDS_SPACE const char*)p;
}

__device__ __forceinline__ unsigned short f32_bf16(float f) {
  unsigned int u = __builtin_bit_cast(unsigned int, f);
  u += 0x7fffu + ((u >> 16) & 1u);   // round-to-nearest-even
  return (unsigned short)(u >> 16);
}
__device__ __forceinline__ unsigned pack_bf16x2(float lo, float hi) {
  return (unsigned)f32_bf16(lo) | ((unsigned)f32_bf16(hi) << 16);
}

// ---- async global->LDS 16B chunk copies (tracked by ASYNCcnt) --------------
// 32-wide tile (64B rows, 4 chunks/row): chunk c in 0..511.
__device__ __forceinline__ void async_chunk32(unsigned ldsBase, const unsigned short* gRow0,
                                              int rowStrideElems, int c) {
  const int row = c >> 2;
  const int within = (c & 3) * 16;  // bytes
  unsigned lds = ldsBase + (unsigned)(row * (LDSTR * 2) + within);
  const char* g = (const char*)gRow0 + (size_t)row * rowStrideElems * 2 + within;
  asm volatile("global_load_async_to_lds_b128 %0, %1, off" :: "v"(lds), "v"(g) : "memory");
}
// 64-wide tile (128B rows, 8 chunks/row): chunk c in 0..1023.
__device__ __forceinline__ void async_chunk64(unsigned ldsBase, const unsigned short* gRow0,
                                              int rowStrideElems, int c) {
  const int row = c >> 3;
  const int within = (c & 7) * 16;  // bytes
  unsigned lds = ldsBase + (unsigned)(row * (LDSTR2 * 2) + within);
  const char* g = (const char*)gRow0 + (size_t)row * rowStrideElems * 2 + within;
  asm volatile("global_load_async_to_lds_b128 %0, %1, off" :: "v"(lds), "v"(g) : "memory");
}
__device__ __forceinline__ void wait_async() {
  asm volatile("s_wait_asynccnt 0" ::: "memory");
}

// ---- bf16 fragment load: 8 aligned u32 pair reads, straight bit_cast -------
// CDNA5 16-bit A/B layout: lanes 0-15 row=lane, K pairs {0..7,16..23};
// lanes 16-31 K pairs {8..15,24..31}.  kOff = K offset within staged tile.
template <int STRIDE>
__device__ __forceinline__ v16bf load_frag(const unsigned short* s, int rowBase, int kOff, int lane) {
  const int half = lane >> 4;
  const unsigned short* r = s + (rowBase + (lane & 15)) * STRIDE + kOff;
  v8u t;
#pragma unroll
  for (int v = 0; v < 8; ++v) {
    const int k = (v < 4) ? (2 * v + 8 * half) : (16 + 2 * (v - 4) + 8 * half);
    t[v] = *(const unsigned int*)(r + k);
  }
  return __builtin_bit_cast(v16bf, t);
}

// ---- stage 16 f32 -> 8 packed bf16 pairs into LDS --------------------------
__device__ __forceinline__ void stage16f(unsigned short* dst, const float4* src) {
  float4 a = src[0], b = src[1], c = src[2], d = src[3];
  unsigned* p = (unsigned*)dst;
  p[0] = pack_bf16x2(a.x, a.y); p[1] = pack_bf16x2(a.z, a.w);
  p[2] = pack_bf16x2(b.x, b.y); p[3] = pack_bf16x2(b.z, b.w);
  p[4] = pack_bf16x2(c.x, c.y); p[5] = pack_bf16x2(c.z, c.w);
  p[6] = pack_bf16x2(d.x, d.y); p[7] = pack_bf16x2(d.z, d.w);
}

// ---------------------------------------------------------------------------
// Streaming f32 -> bf16 conversion (inputs & weights), 8 elems/thread.
// ---------------------------------------------------------------------------
__global__ __launch_bounds__(256) void cvt_bf16(const float* __restrict__ x,
                                                unsigned short* __restrict__ y, int n) {
  const int i = (int)(blockIdx.x * 256 + threadIdx.x) * 8;
  if (i >= n) return;
  float4 a = *(const float4*)(x + i);
  float4 b = *(const float4*)(x + i + 4);
  uint4 o;
  o.x = pack_bf16x2(a.x, a.y);
  o.y = pack_bf16x2(a.z, a.w);
  o.z = pack_bf16x2(b.x, b.y);
  o.w = pack_bf16x2(b.z, b.w);
  *(uint4*)(y + i) = o;
}

// ---------------------------------------------------------------------------
// C[M=4096, DM] = A[M,DM](bf16) * W[DM,DM](bf16)^T + bias(f32).
// Out: bf16 (intermediates) or f32 (final output).  Grid (8, 32), 256 thr.
// Wave tile 32x64 -> 2x4 wmma.  Double-buffered async-LDS staging: next
// K-tile's copy overlaps current tile's ds loads + WMMAs.
// ---------------------------------------------------------------------------
template <bool F32OUT>
__global__ __launch_bounds__(256) void proj_gemm(const unsigned short* __restrict__ A,
                                                 const unsigned short* __restrict__ W,
                                                 const float* __restrict__ bias,
                                                 void* __restrict__ Cv) {
  __shared__ unsigned short As[2][128 * LDSTR];
  __shared__ unsigned short Bs[2][128 * LDSTR];
  const int tid  = threadIdx.x;
  const int lane = tid & 31;
  const int wave = tid >> 5;
  const int wm = wave >> 1, wn = wave & 1;
  const int rowBase = blockIdx.y * 128;
  const int colBase = blockIdx.x * 128;
  const unsigned asOff[2] = {lds_off(As[0]), lds_off(As[1])};
  const unsigned bsOff[2] = {lds_off(Bs[0]), lds_off(Bs[1])};

  v8f acc[2][4];
#pragma unroll
  for (int i = 0; i < 2; ++i)
#pragma unroll
    for (int j = 0; j < 4; ++j) acc[i][j] = 0.0f;

  auto issue = [&](int buf, int k0) {
    const unsigned short* aT = A + (size_t)rowBase * DM + k0;
    const unsigned short* wT = W + (size_t)colBase * DM + k0;
    async_chunk32(asOff[buf], aT, DM, tid);
    async_chunk32(asOff[buf], aT, DM, tid + 256);
    async_chunk32(bsOff[buf], wT, DM, tid);
    async_chunk32(bsOff[buf], wT, DM, tid + 256);
  };

  issue(0, 0);
  constexpr int NIT = DM / 32;
  for (int it = 0; it < NIT; ++it) {
    wait_async();          // my fills of buf[it&1] complete
    __syncthreads();       // everyone's fills done; everyone done reading other buf
    if (it + 1 < NIT) issue((it + 1) & 1, (it + 1) * 32);
    const unsigned short* as = As[it & 1];
    const unsigned short* bs = Bs[it & 1];
    v16bf af[2], bfm[4];
#pragma unroll
    for (int i = 0; i < 2; ++i) af[i]  = load_frag<LDSTR>(as, wm * 32 + i * 16, 0, lane);
#pragma unroll
    for (int j = 0; j < 4; ++j) bfm[j] = load_frag<LDSTR>(bs, wn * 64 + j * 16, 0, lane);
#pragma unroll
    for (int i = 0; i < 2; ++i)
#pragma unroll
      for (int j = 0; j < 4; ++j)
        acc[i][j] = __builtin_amdgcn_wmma_f32_16x16x32_bf16(
            false, af[i], false, bfm[j], (short)0, acc[i][j], false, false);
  }

  // C/D layout: VGPR r -> M = r (+8 for lanes 16-31), lane&15 -> N.
  const int radd = (lane < 16) ? 0 : 8;
  const int cl = lane & 15;
#pragma unroll
  for (int j = 0; j < 4; ++j) {
    const int col = colBase + wn * 64 + j * 16 + cl;
    const float b = bias[col];
#pragma unroll
    for (int i = 0; i < 2; ++i) {
      const int tr = rowBase + wm * 32 + i * 16 + radd;
#pragma unroll
      for (int r = 0; r < 8; ++r) {
        const float v = acc[i][j][r] + b;
        if constexpr (F32OUT)
          ((float*)Cv)[(size_t)(tr + r) * DM + col] = v;
        else
          ((unsigned short*)Cv)[(size_t)(tr + r) * DM + col] = f32_bf16(v);
      }
    }
  }
}

// ---------------------------------------------------------------------------
// logits[bh,q,k] = (Q_h[q,:] . K_h[k,:]) / 8 + mask[b,k] * -1e10  (f32 out)
// Grid (S/128, S/128, B*H).  K-dim = 64: stage the whole 128x64 tile once
// (async), then run both WMMA k-steps from LDS.  One barrier round total.
// ---------------------------------------------------------------------------
__global__ __launch_bounds__(256) void attn_logits(const unsigned short* __restrict__ qb,
                                                   const unsigned short* __restrict__ kb,
                                                   const float* __restrict__ maskp,
                                                   float* __restrict__ attn) {
  __shared__ unsigned short As[128 * LDSTR2];
  __shared__ unsigned short Bs[128 * LDSTR2];
  const int bh = blockIdx.z;
  const int b = bh >> 4, h = bh & 15;
  const unsigned short* Q = qb + (size_t)b * SEQ * DM + h * DH;
  const unsigned short* K = kb + (size_t)b * SEQ * DM + h * DH;
  float* out = attn + (size_t)bh * SEQ * SEQ;
  const float* mrow = maskp + (size_t)b * SEQ;

  const int tid = threadIdx.x, lane = tid & 31, wave = tid >> 5;
  const int wm = wave >> 1, wn = wave & 1;
  const int rowBase = blockIdx.y * 128, colBase = blockIdx.x * 128;
  const unsigned asOff = lds_off(As);
  const unsigned bsOff = lds_off(Bs);

  const unsigned short* qT = Q + (size_t)rowBase * DM;
  const unsigned short* kT = K + (size_t)colBase * DM;
#pragma unroll
  for (int c = 0; c < 4; ++c) {
    async_chunk64(asOff, qT, DM, tid + c * 256);
    async_chunk64(bsOff, kT, DM, tid + c * 256);
  }
  wait_async();
  __syncthreads();

  v8f acc[2][4];
#pragma unroll
  for (int i = 0; i < 2; ++i)
#pragma unroll
    for (int j = 0; j < 4; ++j) acc[i][j] = 0.0f;

#pragma unroll
  for (int k0 = 0; k0 < DH; k0 += 32) {
    v16bf af[2], bfm[4];
#pragma unroll
    for (int i = 0; i < 2; ++i) af[i]  = load_frag<LDSTR2>(As, wm * 32 + i * 16, k0, lane);
#pragma unroll
    for (int j = 0; j < 4; ++j) bfm[j] = load_frag<LDSTR2>(Bs, wn * 64 + j * 16, k0, lane);
#pragma unroll
    for (int i = 0; i < 2; ++i)
#pragma unroll
      for (int j = 0; j < 4; ++j)
        acc[i][j] = __builtin_amdgcn_wmma_f32_16x16x32_bf16(
            false, af[i], false, bfm[j], (short)0, acc[i][j], false, false);
  }

  const int radd = (lane < 16) ? 0 : 8;
  const int cl = lane & 15;
#pragma unroll
  for (int j = 0; j < 4; ++j) {
    const int col = colBase + wn * 64 + j * 16 + cl;
    const float madd = mrow[col] * -1e10f;
#pragma unroll
    for (int i = 0; i < 2; ++i) {
      const int tr = rowBase + wm * 32 + i * 16 + radd;
#pragma unroll
      for (int r = 0; r < 8; ++r)
        out[(size_t)(tr + r) * SEQ + col] = acc[i][j][r] * 0.125f + madd;
    }
  }
}

// ---------------------------------------------------------------------------
// Row softmax over 2048 elements, in place (f32).  Grid (S, B*H), 256 thr.
// Fully coalesced; __expf keeps per-element VALU tiny vs the 23.3 TB/s stream.
// ---------------------------------------------------------------------------
__global__ __launch_bounds__(256) void softmax_rows(float* __restrict__ attn) {
  float* row = attn + ((size_t)blockIdx.y * SEQ + blockIdx.x) * SEQ;
  const int tid = threadIdx.x, lane = tid & 31, wave = tid >> 5;
  __shared__ float smax[8], ssum[8];

  float v[8];
  float m = -3.4e38f;
#pragma unroll
  for (int i = 0; i < 8; ++i) { v[i] = row[tid + i * 256]; m = fmaxf(m, v[i]); }
#pragma unroll
  for (int off = 16; off > 0; off >>= 1) m = fmaxf(m, __shfl_xor(m, off, 32));
  if (lane == 0) smax[wave] = m;
  __syncthreads();
  m = smax[0];
#pragma unroll
  for (int w = 1; w < 8; ++w) m = fmaxf(m, smax[w]);

  float s = 0.0f;
#pragma unroll
  for (int i = 0; i < 8; ++i) { v[i] = __expf(v[i] - m); s += v[i]; }
#pragma unroll
  for (int off = 16; off > 0; off >>= 1) s += __shfl_xor(s, off, 32);
  if (lane == 0) ssum[wave] = s;
  __syncthreads();
  s = 0.0f;
#pragma unroll
  for (int w = 0; w < 8; ++w) s += ssum[w];
  const float inv = 1.0f / s;
#pragma unroll
  for (int i = 0; i < 8; ++i) row[tid + i * 256] = v[i] * inv;
}

// ---------------------------------------------------------------------------
// ctx[bh,q,d] = sum_k P[bh,q,k] * V_h[k,d] -> concat[B,S,H*DH] (bf16).
// P is f32 (softmax output), converted once while staging; V is bf16.
// Grid (S/128, B*H).  Block tile 128x64, wave tile 32x32 -> 2x2 wmma.
// ---------------------------------------------------------------------------
__global__ __launch_bounds__(256) void attn_ctx(const float* __restrict__ attn,
                                                const unsigned short* __restrict__ vb,
                                                unsigned short* __restrict__ concat) {
  __shared__ unsigned short As[128 * LDSTR];
  __shared__ unsigned short Bs[64 * LDSTR];
  const int bh = blockIdx.y, b = bh >> 4, h = bh & 15;
  const float* P = attn + (size_t)bh * SEQ * SEQ;
  const unsigned short* V = vb + (size_t)b * SEQ * DM + h * DH;

  const int tid = threadIdx.x, lane = tid & 31, wave = tid >> 5;
  const int wm = wave >> 1, wn = wave & 1;
  const int rowBase = blockIdx.x * 128;

  v8f acc[2][2];
#pragma unroll
  for (int i = 0; i < 2; ++i)
#pragma unroll
    for (int j = 0; j < 2; ++j) acc[i][j] = 0.0f;

  const int lr = tid >> 1, lc = (tid & 1) * 16;
  const int vd = tid & 63, vk0 = (tid >> 6) * 8;

  for (int k0 = 0; k0 < SEQ; k0 += 32) {
    stage16f(As + lr * LDSTR + lc, (const float4*)(P + (size_t)(rowBase + lr) * SEQ + k0 + lc));
    // B role: W[d,k] = V[k,d] (bf16 transpose-gather, stride DM)
#pragma unroll
    for (int kk = 0; kk < 8; ++kk)
      Bs[vd * LDSTR + vk0 + kk] = V[(size_t)(k0 + vk0 + kk) * DM + vd];
    __syncthreads();
    v16bf af[2], bfm[2];
#pragma unroll
    for (int i = 0; i < 2; ++i) af[i]  = load_frag<LDSTR>(As, wm * 32 + i * 16, 0, lane);
#pragma unroll
    for (int j = 0; j < 2; ++j) bfm[j] = load_frag<LDSTR>(Bs, wn * 32 + j * 16, 0, lane);
#pragma unroll
    for (int i = 0; i < 2; ++i)
#pragma unroll
      for (int j = 0; j < 2; ++j)
        acc[i][j] = __builtin_amdgcn_wmma_f32_16x16x32_bf16(
            false, af[i], false, bfm[j], (short)0, acc[i][j], false, false);
    __syncthreads();
  }

  const int radd = (lane < 16) ? 0 : 8;
  const int cl = lane & 15;
#pragma unroll
  for (int j = 0; j < 2; ++j) {
    const int col = wn * 32 + j * 16 + cl;   // 0..63 (depth)
#pragma unroll
    for (int i = 0; i < 2; ++i) {
      const int tr = rowBase + wm * 32 + i * 16 + radd;
#pragma unroll
      for (int r = 0; r < 8; ++r)
        concat[((size_t)b * SEQ + tr + r) * DM + h * DH + col] = f32_bf16(acc[i][j][r]);
    }
  }
}

// ---------------------------------------------------------------------------
extern "C" void kernel_launch(void* const* d_in, const int* in_sizes, int n_in,
                              void* d_out, int out_size, void* d_ws, size_t ws_size,
                              hipStream_t stream) {
  (void)in_sizes; (void)n_in; (void)out_size; (void)ws_size;
  const float* value = (const float*)d_in[0];
  const float* keyi  = (const float*)d_in[1];
  const float* query = (const float*)d_in[2];
  const float* mask  = (const float*)d_in[3];
  const float* Wq = (const float*)d_in[4];
  const float* bq = (const float*)d_in[5];
  const float* Wk = (const float*)d_in[6];
  const float* bk = (const float*)d_in[7];
  const float* Wv = (const float*)d_in[8];
  const float* bv = (const float*)d_in[9];
  const float* Wo = (const float*)d_in[10];
  const float* bo = (const float*)d_in[11];

  float* outp = (float*)d_out;
  float* attn = outp + (size_t)BATCH * SEQ * DM;   // second tuple element

  const size_t actE = (size_t)BATCH * SEQ * DM;    // 4M elems
  const size_t wE   = (size_t)DM * DM;             // 1M elems
  unsigned short* w = (unsigned short*)d_ws;       // 64 MB total bf16
  unsigned short* xq  = w;
  unsigned short* xk  = xq + actE;
  unsigned short* xv  = xk + actE;
  unsigned short* wqb = xv + actE;
  unsigned short* wkb = wqb + wE;
  unsigned short* wvb = wkb + wE;
  unsigned short* wob = wvb + wE;
  unsigned short* qb  = wob + wE;
  unsigned short* kb  = qb + actE;
  unsigned short* vb2 = kb + actE;
  unsigned short* cb  = vb2 + actE;

  dim3 blk(256);
  cvt_bf16<<<dim3((int)(actE / 2048)), blk, 0, stream>>>(query, xq, (int)actE);
  cvt_bf16<<<dim3((int)(actE / 2048)), blk, 0, stream>>>(keyi,  xk, (int)actE);
  cvt_bf16<<<dim3((int)(actE / 2048)), blk, 0, stream>>>(value, xv, (int)actE);
  cvt_bf16<<<dim3((int)(wE / 2048)),   blk, 0, stream>>>(Wq, wqb, (int)wE);
  cvt_bf16<<<dim3((int)(wE / 2048)),   blk, 0, stream>>>(Wk, wkb, (int)wE);
  cvt_bf16<<<dim3((int)(wE / 2048)),   blk, 0, stream>>>(Wv, wvb, (int)wE);
  cvt_bf16<<<dim3((int)(wE / 2048)),   blk, 0, stream>>>(Wo, wob, (int)wE);

  dim3 gProj(DM / 128, (BATCH * SEQ) / 128);       // 8 x 32
  proj_gemm<false><<<gProj, blk, 0, stream>>>(xq, wqb, bq, qb);
  proj_gemm<false><<<gProj, blk, 0, stream>>>(xk, wkb, bk, kb);
  proj_gemm<false><<<gProj, blk, 0, stream>>>(xv, wvb, bv, vb2);

  attn_logits<<<dim3(SEQ / 128, SEQ / 128, BATCH * NH), blk, 0, stream>>>(qb, kb, mask, attn);
  softmax_rows<<<dim3(SEQ, BATCH * NH), blk, 0, stream>>>(attn);
  attn_ctx<<<dim3(SEQ / 128, BATCH * NH), blk, 0, stream>>>(attn, vb2, cb);

  proj_gemm<true><<<gProj, blk, 0, stream>>>(cb, wob, bo, outp);
}
